// MultiHeadLatentAttention_55594056679625
// MI455X (gfx1250) — matrix operand order repaired and checked by reading, not compiled, not run on hardware
//
#include <hip/hip_runtime.h>
#include <hip/hip_bf16.h>
#include <math.h>

typedef __attribute__((ext_vector_type(16))) _Float16 v16h;
typedef __attribute__((ext_vector_type(2)))  _Float16 v2h;
typedef __attribute__((ext_vector_type(8)))  float    v8f;
typedef __attribute__((ext_vector_type(2)))  float    vf2;

#define D_MODEL  1024
#define N_HEADS  16
#define HEAD_DIM 64
#define KV_DIM   256
#define SEQ_B    2
#define SEQ_T    2048
#define ROWS     (SEQ_B*SEQ_T)   // 4096
#define PADW     36              // padded LDS row (halves): 72B, conflict-free across 16 rows

static __device__ __forceinline__ v8f wmma16x16x32(v16h a, v16h b, v8f c) {
  // D = A(16x32 f16) * B(32x16 f16) + C(16x16 f32)
  return __builtin_amdgcn_wmma_f32_16x16x32_f16(false, a, false, b, (short)0, c, false, false);
}

// ---------------- weight convert + transpose: W[K,N] f32 -> Wt[N,K] f16 ----------------
__global__ void wcvt_transpose(const float* __restrict__ W, _Float16* __restrict__ Wt,
                               int K, int N) {
  int tid = blockIdx.x * blockDim.x + threadIdx.x;
  int total = K * N;
  if (tid >= total) return;
  int n = tid % N, k = tid / N;
  Wt[(size_t)n * K + k] = (_Float16)W[(size_t)k * N + n];
}

// ---------------- generic GEMM: C[M,N] = A[M,K](f32) @ Wt[N,K](f16)^T + bias ----------------
// one wave computes a 16x64 strip (4 accumulators): A-fragment reused across 4 WMMAs per K-chunk
__global__ void __launch_bounds__(256) gemm_aF32_wT16(
    const float* __restrict__ A, const _Float16* __restrict__ Wt,
    const float* __restrict__ bias, float* __restrict__ C,
    int M, int N, int K) {
  int wave = blockIdx.x * (blockDim.x >> 5) + (threadIdx.x >> 5);
  int lane = threadIdx.x & 31;
  int h = lane >> 4, idx = lane & 15;
  int nstrips = N >> 6;                       // 64-wide output strips
  int mt = wave / nstrips, ns = wave % nstrips;
  if (mt >= (M >> 4)) return;

  const float*    arow  = A  + (size_t)(mt * 16 + idx) * K;           // A-frag: lane = row
  const _Float16* wbase = Wt + (size_t)(ns * 64 + idx) * K;  // B-frag: lane = col
  v8f acc[4] = {{}, {}, {}, {}};
  for (int kb = 0; kb < K; kb += 32) {
    if (kb + 32 < K) __builtin_prefetch(arow + kb + 32, 0, 0);        // global_prefetch_b8
    v16h a;
#pragma unroll
    for (int v = 0; v < 8; ++v) {
      // 16-bit A 16x32 layout: VGPR v -> K = (v>=4?16:0) + h*8 + (v&3)*2 + {0,1}
      int koff = ((v & 4) ? 16 : 0) + h * 8 + (v & 3) * 2;
      vf2 f = *(const vf2*)(arow + kb + koff);
      a[2 * v]     = (_Float16)f.x;
      a[2 * v + 1] = (_Float16)f.y;
    }
#pragma unroll
    for (int n0 = 0; n0 < 4; ++n0) {
      const _Float16* wrow = wbase + (size_t)(n0 * 16) * K + kb;
      v16h b;
#pragma unroll
      for (int v = 0; v < 8; ++v) {
        // 16-bit B 32x16 layout: VGPR v -> K = (v>=4?16:0) + h*8 + (v&3)*2 + {0,1}
        v2h p = *(const v2h*)(wrow + ((v & 4) ? 16 : 0) + h * 8 + (v & 3) * 2);
        b[2 * v]     = p.x;
        b[2 * v + 1] = p.y;
      }
      acc[n0] = wmma16x16x32(a, b, acc[n0]);
    }
  }
#pragma unroll
  for (int n0 = 0; n0 < 4; ++n0) {
    int col = ns * 64 + n0 * 16 + idx;
    float bb = bias ? bias[col] : 0.0f;
#pragma unroll
    for (int r = 0; r < 8; ++r) {            // C layout: row = r + 8*h, col = lane&15
      int row = mt * 16 + r + 8 * h;
      C[(size_t)row * N + col] = acc[n0][r] + bb;
    }
  }
}

// ---------------- RoPE on q: q_lin[B,T,H,64] f32 -> qh[B,H,T,64] f16 ----------------
__global__ void rope_q(const float* __restrict__ qlin, _Float16* __restrict__ qh) {
  int tid = blockIdx.x * blockDim.x + threadIdx.x;
  const int total = SEQ_B * SEQ_T * N_HEADS * 32;
  if (tid >= total) return;
  int d = tid & 31;
  int r = tid >> 5;
  int head = r & 15; r >>= 4;
  int t = r & (SEQ_T - 1); int b = r >> 11;
  const float LOG_THETA = 9.210340371976184f;      // ln(10000)
  float inv = __expf(-(float)d * (LOG_THETA / 32.0f));
  float ang = (float)t * inv;
  float cs = __cosf(ang), sn = __sinf(ang);
  size_t bi = ((size_t)(b * SEQ_T + t) * N_HEADS + head) * HEAD_DIM;
  float q0 = qlin[bi + d];
  float q1 = qlin[bi + d + 32];
  size_t bo = ((size_t)(b * N_HEADS + head) * SEQ_T + t) * HEAD_DIM;
  qh[bo + d]      = (_Float16)(q0 * cs - q1 * sn);   // q*cos + rotate_half(q)*sin
  qh[bo + d + 32] = (_Float16)(q1 * cs + q0 * sn);
}

// ---------------- k path: (k_lin * t) @ Wkr : per-head [T,64]@[64,64], WMMA ----------------
__global__ void __launch_bounds__(256) krot(
    const float* __restrict__ klin /*[B,T,H,64]*/, const _Float16* __restrict__ Wkrt /*[64,64] n-major*/,
    _Float16* __restrict__ kh /*[B,H,T,64]*/) {
  int wave = blockIdx.x * (blockDim.x >> 5) + (threadIdx.x >> 5);
  int lane = threadIdx.x & 31;
  int h = lane >> 4, idx = lane & 15;
  int ttile = wave & (SEQ_T / 16 - 1);
  int bh = wave >> 7;
  if (bh >= SEQ_B * N_HEADS) return;
  int b = bh >> 4, head = bh & 15;
  int trow = ttile * 16 + idx;
  const float* arow = klin + ((size_t)(b * SEQ_T + trow) * N_HEADS + head) * HEAD_DIM;
  float tscale = (float)trow;
  v8f acc[4] = {{}, {}, {}, {}};
  for (int kb = 0; kb < 64; kb += 32) {
    v16h a;
#pragma unroll
    for (int v = 0; v < 8; ++v) {
      int koff = ((v & 4) ? 16 : 0) + h * 8 + (v & 3) * 2;
      vf2 f = *(const vf2*)(arow + kb + koff);
      a[2 * v]     = (_Float16)(f.x * tscale);
      a[2 * v + 1] = (_Float16)(f.y * tscale);
    }
#pragma unroll
    for (int n0 = 0; n0 < 4; ++n0) {
      const _Float16* wrow = Wkrt + (size_t)(n0 * 16 + idx) * 64 + kb;
      v16h bf;
#pragma unroll
      for (int v = 0; v < 8; ++v) {
        v2h p = *(const v2h*)(wrow + ((v & 4) ? 16 : 0) + h * 8 + (v & 3) * 2);
        bf[2 * v] = p.x; bf[2 * v + 1] = p.y;
      }
      acc[n0] = wmma16x16x32(a, bf, acc[n0]);
    }
  }
  size_t obase = (size_t)bh * SEQ_T * HEAD_DIM;
#pragma unroll
  for (int n0 = 0; n0 < 4; ++n0)
#pragma unroll
    for (int r = 0; r < 8; ++r) {
      int row = ttile * 16 + r + 8 * h;
      kh[obase + (size_t)row * HEAD_DIM + n0 * 16 + idx] = (_Float16)acc[n0][r];
    }
}

// ---------------- v pack: v_lin[B,T,H,64] f32 -> vT[B,H,64,T] f16 ----------------
__global__ void vpack(const float* __restrict__ vlin, _Float16* __restrict__ vt) {
  int tid = blockIdx.x * blockDim.x + threadIdx.x;
  const int total = SEQ_B * SEQ_T * N_HEADS * HEAD_DIM;
  if (tid >= total) return;
  int d = tid & 63; int r = tid >> 6;
  int head = r & 15; r >>= 4;
  int t = r & (SEQ_T - 1); int b = r >> 11;
  float val = vlin[((size_t)(b * SEQ_T + t) * N_HEADS + head) * HEAD_DIM + d];
  vt[(((size_t)(b * N_HEADS + head) * HEAD_DIM) + d) * SEQ_T + t] = (_Float16)val;
}

// ---------------- flash attention: per wave 16 query rows, 32-key tiles, online softmax ----------------
__global__ void __launch_bounds__(256) flash_attn(
    const _Float16* __restrict__ qh, const _Float16* __restrict__ kh,
    const _Float16* __restrict__ vt, float* __restrict__ y) {
  __shared__ _Float16 plds[8][16 * PADW];   // per-wave P repack tile (C-layout -> A-layout)
  int waveId = threadIdx.x >> 5;
  int lane = threadIdx.x & 31;
  int h = lane >> 4, idx = lane & 15;
  int rowblock = blockIdx.x & 15;           // T/128 = 16 row blocks
  int bh = blockIdx.x >> 4;
  int b = bh >> 4, head = bh & 15;
  int qbase = rowblock * 128 + waveId * 16;

  const _Float16* qP = qh + (size_t)bh * SEQ_T * HEAD_DIM;
  const _Float16* kP = kh + (size_t)bh * SEQ_T * HEAD_DIM;
  const _Float16* vP = vt + (size_t)bh * HEAD_DIM * SEQ_T;

  v16h aq[2];                               // Q 16x64 as two 16x32 A-fragments
#pragma unroll
  for (int f = 0; f < 2; ++f) {
    const _Float16* qrow = qP + (size_t)(qbase + idx) * HEAD_DIM + f * 32;
#pragma unroll
    for (int v = 0; v < 8; ++v) {
      int koff = ((v & 4) ? 16 : 0) + h * 8 + (v & 3) * 2;
      v2h p = *(const v2h*)(qrow + koff);
      aq[f][2 * v] = p.x; aq[f][2 * v + 1] = p.y;
    }
  }

  v8f acc[4] = {{}, {}, {}, {}};            // O accumulator 16x64 (4 dim-subtiles)
  float mrow[8], lrow[8];
#pragma unroll
  for (int r = 0; r < 8; ++r) { mrow[r] = -1e30f; lrow[r] = 0.0f; }

  _Float16* myP = &plds[waveId][0];
  int jend = qbase + 15;

  for (int jb = 0; jb <= jend; jb += 32) {
    if (jb + 32 <= jend) {                  // prefetch next key tile (global_prefetch_b8)
      __builtin_prefetch(kP + (size_t)(jb + 32 + idx) * HEAD_DIM, 0, 0);
      __builtin_prefetch(vP + (size_t)idx * SEQ_T + jb + 32, 0, 0);
    }
    v8f S[2];
#pragma unroll
    for (int sub = 0; sub < 2; ++sub) {     // S = Q @ K^T, keys jb+sub*16 .. +15
      v8f s = {};
#pragma unroll
      for (int f = 0; f < 2; ++f) {
        const _Float16* krow = kP + (size_t)(jb + sub * 16 + idx) * HEAD_DIM + f * 32;
        v16h bk;
#pragma unroll
        for (int v = 0; v < 8; ++v) {
          v2h p = *(const v2h*)(krow + ((v & 4) ? 16 : 0) + h * 8 + (v & 3) * 2);
          bk[2 * v] = p.x; bk[2 * v + 1] = p.y;
        }
        s = wmma16x16x32(aq[f], bk, s);
      }
      S[sub] = s;
    }
    // scale 1/sqrt(64) + causal mask
#pragma unroll
    for (int sub = 0; sub < 2; ++sub) {
      int key = jb + sub * 16 + idx;
#pragma unroll
      for (int r = 0; r < 8; ++r) {
        int qr = qbase + r + 8 * h;
        float sv = S[sub][r] * 0.125f;
        S[sub][r] = (key > qr) ? -1e30f : sv;
      }
    }
    // online softmax: row max (xor-shuffle within 16-lane column group)
    float mnew[8], alpha[8];
#pragma unroll
    for (int r = 0; r < 8; ++r) {
      float mx = fmaxf(S[0][r], S[1][r]);
#pragma unroll
      for (int sh = 1; sh <= 8; sh <<= 1)
        mx = fmaxf(mx, __shfl_xor(mx, sh, 32));
      mnew[r]  = fmaxf(mrow[r], mx);
      alpha[r] = __expf(mrow[r] - mnew[r]);
      mrow[r]  = mnew[r];
    }
    // P = exp(S - m); store to LDS in C-layout; accumulate row sums
#pragma unroll
    for (int r = 0; r < 8; ++r) {
      float p0 = __expf(S[0][r] - mnew[r]);
      float p1 = __expf(S[1][r] - mnew[r]);
      myP[(r + 8 * h) * PADW + idx]      = (_Float16)p0;
      myP[(r + 8 * h) * PADW + 16 + idx] = (_Float16)p1;
      float s = p0 + p1;
#pragma unroll
      for (int sh = 1; sh <= 8; sh <<= 1)
        s += __shfl_xor(s, sh, 32);
      lrow[r] = lrow[r] * alpha[r] + s;
    }
#pragma unroll
    for (int d = 0; d < 4; ++d)
#pragma unroll
      for (int r = 0; r < 8; ++r)
        acc[d][r] *= alpha[r];
    // reload P as A-fragment (16x32) from LDS (wave-private; DS ops in-order per wave)
    v16h ap;
#pragma unroll
    for (int v = 0; v < 8; ++v) {
      int koff = ((v & 4) ? 16 : 0) + h * 8 + (v & 3) * 2;
      v2h p = *(const v2h*)(myP + idx * PADW + koff);
      ap[2 * v] = p.x; ap[2 * v + 1] = p.y;
    }
    // acc += P @ V (V fragments from [dim,key]-major vT => contiguous key pairs)
#pragma unroll
    for (int d = 0; d < 4; ++d) {
      const _Float16* vrow = vP + (size_t)(d * 16 + idx) * SEQ_T + jb;
      v16h bv;
#pragma unroll
      for (int v = 0; v < 8; ++v) {
        v2h p = *(const v2h*)(vrow + ((v & 4) ? 16 : 0) + h * 8 + (v & 3) * 2);
        bv[2 * v] = p.x; bv[2 * v + 1] = p.y;
      }
      acc[d] = wmma16x16x32(ap, bv, acc[d]);
    }
  }
  // normalize and write y as [B,T,D_MODEL] (heads concatenated) for the Wo GEMM
#pragma unroll
  for (int d = 0; d < 4; ++d)
#pragma unroll
    for (int r = 0; r < 8; ++r) {
      int row = qbase + r + 8 * h;
      y[(size_t)(b * SEQ_T + row) * D_MODEL + head * HEAD_DIM + d * 16 + idx] =
          acc[d][r] / lrow[r];
    }
}

// ------------------------------------------------------------------------------------
extern "C" void kernel_launch(void* const* d_in, const int* in_sizes, int n_in,
                              void* d_out, int out_size, void* d_ws, size_t ws_size,
                              hipStream_t stream) {
  (void)in_sizes; (void)n_in; (void)out_size; (void)ws_size;
  const float* x   = (const float*)d_in[0];
  // d_in[1] = causal mask, handled analytically
  const float* Wq  = (const float*)d_in[2];
  const float* bq  = (const float*)d_in[3];
  const float* Wkv = (const float*)d_in[4];
  const float* bkv = (const float*)d_in[5];
  const float* Wk  = (const float*)d_in[6];
  const float* bk  = (const float*)d_in[7];
  const float* Wv  = (const float*)d_in[8];
  const float* bv  = (const float*)d_in[9];
  const float* Wo  = (const float*)d_in[10];
  const float* bo  = (const float*)d_in[11];
  const float* Wkr = (const float*)d_in[12];
  float* out = (float*)d_out;

  char* ws = (char*)d_ws;
  size_t off = 0;
  auto alloc = [&](size_t bytes) -> void* {
    void* p = ws + off;
    off = (off + bytes + 255) & ~(size_t)255;
    return p;
  };
  _Float16* wtq  = (_Float16*)alloc((size_t)D_MODEL * D_MODEL * 2);
  _Float16* wtkv = (_Float16*)alloc((size_t)D_MODEL * KV_DIM * 2);
  _Float16* wtk  = (_Float16*)alloc((size_t)KV_DIM * D_MODEL * 2);
  _Float16* wtv  = (_Float16*)alloc((size_t)KV_DIM * D_MODEL * 2);
  _Float16* wto  = (_Float16*)alloc((size_t)D_MODEL * D_MODEL * 2);
  _Float16* wtkr = (_Float16*)alloc((size_t)HEAD_DIM * HEAD_DIM * 2);
  float*    kv   = (float*)   alloc((size_t)ROWS * KV_DIM * 4);
  float*    qlin = (float*)   alloc((size_t)ROWS * D_MODEL * 4);
  float*    klin = (float*)   alloc((size_t)ROWS * D_MODEL * 4);
  float*    vlin = (float*)   alloc((size_t)ROWS * D_MODEL * 4);
  _Float16* qhh  = (_Float16*)alloc((size_t)ROWS * D_MODEL * 2);
  _Float16* khh  = (_Float16*)alloc((size_t)ROWS * D_MODEL * 2);
  _Float16* vth  = (_Float16*)alloc((size_t)ROWS * D_MODEL * 2);
  float*    y    = qlin;   // q_lin is dead after rope_q -> reuse for attention output

  const int TPB = 256;
  // 1) weight transposes to f16 [N,K]
  wcvt_transpose<<<(D_MODEL * D_MODEL + TPB - 1) / TPB, TPB, 0, stream>>>(Wq,  wtq,  D_MODEL, D_MODEL);
  wcvt_transpose<<<(D_MODEL * KV_DIM  + TPB - 1) / TPB, TPB, 0, stream>>>(Wkv, wtkv, D_MODEL, KV_DIM);
  wcvt_transpose<<<(KV_DIM  * D_MODEL + TPB - 1) / TPB, TPB, 0, stream>>>(Wk,  wtk,  KV_DIM,  D_MODEL);
  wcvt_transpose<<<(KV_DIM  * D_MODEL + TPB - 1) / TPB, TPB, 0, stream>>>(Wv,  wtv,  KV_DIM,  D_MODEL);
  wcvt_transpose<<<(D_MODEL * D_MODEL + TPB - 1) / TPB, TPB, 0, stream>>>(Wo,  wto,  D_MODEL, D_MODEL);
  wcvt_transpose<<<(HEAD_DIM * HEAD_DIM + TPB - 1) / TPB, TPB, 0, stream>>>(Wkr, wtkr, HEAD_DIM, HEAD_DIM);

  // 2) projections (each wave = one 16x64 strip; 8 waves/block)
  auto gemm_blocks = [](int M, int N) { return ((M / 16) * (N / 64)) / 8; };
  gemm_aF32_wT16<<<gemm_blocks(ROWS, KV_DIM),  TPB, 0, stream>>>(x,  wtkv, bkv, kv,   ROWS, KV_DIM,  D_MODEL);
  gemm_aF32_wT16<<<gemm_blocks(ROWS, D_MODEL), TPB, 0, stream>>>(x,  wtq,  bq,  qlin, ROWS, D_MODEL, D_MODEL);
  gemm_aF32_wT16<<<gemm_blocks(ROWS, D_MODEL), TPB, 0, stream>>>(kv, wtk,  bk,  klin, ROWS, D_MODEL, KV_DIM);
  gemm_aF32_wT16<<<gemm_blocks(ROWS, D_MODEL), TPB, 0, stream>>>(kv, wtv,  bv,  vlin, ROWS, D_MODEL, KV_DIM);

  // 3) RoPE(q), position-scale + Wkr for k, v transpose-pack
  rope_q<<<(SEQ_B * SEQ_T * N_HEADS * 32 + TPB - 1) / TPB, TPB, 0, stream>>>(qlin, qhh);
  krot<<<(SEQ_B * N_HEADS * (SEQ_T / 16)) / 8, TPB, 0, stream>>>(klin, wtkr, khh);
  vpack<<<(SEQ_B * SEQ_T * N_HEADS * HEAD_DIM + TPB - 1) / TPB, TPB, 0, stream>>>(vlin, vth);

  // 4) flash attention: 8 waves/block * 16 rows/wave = 128 query rows per block
  flash_attn<<<SEQ_B * N_HEADS * (SEQ_T / 128), TPB, 0, stream>>>(qhh, khh, vth, y);

  // 5) output projection
  gemm_aF32_wT16<<<gemm_blocks(ROWS, D_MODEL), TPB, 0, stream>>>(y, wto, bo, out, ROWS, D_MODEL, D_MODEL);
}
